// Transducer_53085795778972
// MI455X (gfx1250) — matrix operand order repaired and checked by reading, not compiled, not run on hardware
//
#include <hip/hip_runtime.h>
#include <math.h>

#define Bn 4
#define Tn 300
#define Un 80
#define Vn 1024
#define DEn 320
#define DINn 640
#define IN_ 512

typedef __bf16 bf16_t;
typedef bf16_t v16bf __attribute__((ext_vector_type(16)));
typedef bf16_t v8bf  __attribute__((ext_vector_type(8)));
typedef float  v8f   __attribute__((ext_vector_type(8)));

union V16U { v16bf v; v8bf h2[2]; };

// ---------------- Kernel 0: Wp -> bf16 ----------------
__global__ void k_wp_bf16(const float* __restrict__ Wp, bf16_t* __restrict__ WpB) {
  int i = blockIdx.x * 256 + threadIdx.x;   // grid sized exactly V*IN/256
  WpB[i] = (bf16_t)Wp[i];
}

// ---------------- Kernel 1: projections ----------------
// encP[(b*T+t)*512+i] = enc(b,t,:) . Wf[i,0:320]
// decP[(b*81+u)*512+i] = bf[i] + dec(b,u,:) . Wf[i,320:640]
__global__ void k_proj(const float* __restrict__ enc, const float* __restrict__ dec,
                       const float* __restrict__ Wf, const float* __restrict__ bfv,
                       float* __restrict__ encP, float* __restrict__ decP) {
  int gid = blockIdx.x * 256 + threadIdx.x;
  const int encTot = (Bn * Tn) * IN_;
  int row = gid >> 9;
  int i   = gid & (IN_ - 1);
  if (gid < encTot) {
    const float* s = enc + row * DEn;
    const float* w = Wf + i * DINn;
    float acc = 0.f;
#pragma unroll 4
    for (int d = 0; d < DEn; ++d) acc += s[d] * w[d];
    encP[gid] = acc;
  } else {
    int rd = row - Bn * Tn;               // 0..323 = b*81+u
    const float* s = dec + rd * DEn;
    const float* w = Wf + i * DINn + DEn;
    float acc = bfv[i];
#pragma unroll 4
    for (int d = 0; d < DEn; ++d) acc += s[d] * w[d];
    decP[rd * IN_ + i] = acc;
  }
}

// ---------------- Kernel 2: fused joint + log-softmax stats ----------------
// grid: (B*T)*3 blocks, 256 threads. Each block: one (b,t), 32 u-cells.
// LDS layout (bytes):
//   0      : h tile   32 rows x 520 bf16 (row stride 1040 B)   = 33280
//   33280  : encRow   512 f32                                  = 2048
//   35328  : bpLds    1024 f32                                 = 4096
//   39424  : cellSum  32 f32
//   39552  : blankL   32 f32
//   39680  : emitL    32 f32
//   39808  : tgtL     32 i32        total = 39936
__global__ __launch_bounds__(256)
void k_joint(const float* __restrict__ encP, const float* __restrict__ decP,
             const bf16_t* __restrict__ WpB, const float* __restrict__ Wp,
             const float* __restrict__ bp, const int* __restrict__ tgt,
             float* __restrict__ blankOut, float* __restrict__ emitOut) {
  extern __shared__ char smem[];
  float* encRow = (float*)(smem + 33280);
  float* bpL    = (float*)(smem + 35328);
  float* cellS  = (float*)(smem + 39424);
  float* blankL = (float*)(smem + 39552);
  float* emitL  = (float*)(smem + 39680);
  int*   tgtL   = (int*)  (smem + 39808);

  const int tid = threadIdx.x;
  const int bt  = blockIdx.x / 3;
  const int ub  = blockIdx.x % 3;
  const int b   = bt / Tn;
  const int u0  = ub * 32;

  for (int i = tid; i < IN_; i += 256) encRow[i] = encP[bt * IN_ + i];
  for (int i = tid; i < Vn;  i += 256) bpL[i] = bp[i];
  if (tid < 32) {
    cellS[tid] = 0.f;
    int u = u0 + tid;
    tgtL[tid] = (u < Un) ? tgt[b * Un + u] : -1;
  }
  __syncthreads();

  // h = tanh(enc_p + dec_p(+bf)) -> bf16 LDS tile
  bf16_t* hT = (bf16_t*)smem;
  for (int idx = tid; idx < 32 * IN_; idx += 256) {
    int m = idx >> 9, k = idx & (IN_ - 1);
    int u = u0 + m; if (u > Un) u = Un;   // clamp; masked cells never written out
    float hv = tanhf(encRow[k] + decP[(b * (Un + 1) + u) * IN_ + k]);
    hT[m * 520 + k] = (bf16_t)hv;
  }
  __syncthreads();

  const int wave = tid >> 5, lane = tid & 31;
  const int hf = lane >> 4, nlo = lane & 15;
  const char* hB = (const char*)smem;

  float sacc0[8], sacc1[8];
#pragma unroll
  for (int r = 0; r < 8; ++r) { sacc0[r] = 0.f; sacc1[r] = 0.f; }

  // 64 N-tiles of 16 vocab cols; wave w handles tiles {w, w+8, ...}
  for (int j = 0; j < 8; ++j) {
    const int n0 = (wave + 8 * j) * 16;
    const int vc = n0 + nlo;
    const bf16_t* bptr = WpB + vc * IN_ + 16 * hf;
    v8f c0 = {}; v8f c1 = {};
#pragma unroll 4
    for (int k0 = 0; k0 < IN_; k0 += 32) {
      v16bf bfrag = *(const v16bf*)(bptr + k0);
      V16U a0, a1;
      int ao = nlo * 1040 + 2 * k0 + 32 * hf;
      a0.h2[0] = *(const v8bf*)(hB + ao);
      a0.h2[1] = *(const v8bf*)(hB + ao + 16);
      a1.h2[0] = *(const v8bf*)(hB + ao + 16 * 1040);
      a1.h2[1] = *(const v8bf*)(hB + ao + 16 * 1040 + 16);
      c0 = __builtin_amdgcn_wmma_f32_16x16x32_bf16(false, a0.v, false, bfrag,
                                                   (short)0, c0, false, false);
      c1 = __builtin_amdgcn_wmma_f32_16x16x32_bf16(false, a1.v, false, bfrag,
                                                   (short)0, c1, false, false);
    }
    float bias = bpL[vc];
#pragma unroll
    for (int r = 0; r < 8; ++r) {
      sacc0[r] += __expf(c0[r] + bias);
      sacc1[r] += __expf(c1[r] + bias);
    }
  }

  // reduce lane partials across each 16-lane half, then across waves via LDS atomics
#pragma unroll
  for (int r = 0; r < 8; ++r) {
    float s0 = sacc0[r], s1 = sacc1[r];
    for (int off = 1; off <= 8; off <<= 1) {
      s0 += __shfl_xor(s0, off, 32);
      s1 += __shfl_xor(s1, off, 32);
    }
    if (nlo == r) {
      atomicAdd(&cellS[r + 8 * hf], s0);           // M-tile 0: cells 0..15
      atomicAdd(&cellS[16 + r + 8 * hf], s1);      // M-tile 1: cells 16..31
    }
  }
  __syncthreads();

  // fp32 side pass: blank (v=0) and target-column logits; 64 wave-tasks
  for (int s = 0; s < 8; ++s) {
    int task = wave * 8 + s;
    int m, row; bool valid; float* dst;
    if (task < 32) { m = task;      row = 0;       valid = (u0 + m) <= Un; dst = blankL + m; }
    else           { m = task - 32; row = tgtL[m]; valid = (row >= 0);     dst = emitL + m;  }
    int rr = valid ? row : 0;
    const bf16_t* hp = (const bf16_t*)smem + m * 520;
    const float* wrow = Wp + rr * IN_;
    float acc = 0.f;
    for (int k = lane; k < IN_; k += 32) acc += (float)hp[k] * wrow[k];
    for (int off = 1; off < 32; off <<= 1) acc += __shfl_xor(acc, off, 32);
    if (lane == 0 && valid) *dst = acc + bpL[rr];
  }
  __syncthreads();

  if (tid < 32) {
    int u = u0 + tid;
    if (u <= Un) {
      float lse = logf(cellS[tid]);
      blankOut[bt * (Un + 1) + u] = blankL[tid] - lse;
      if (u < Un) emitOut[bt * Un + u] = emitL[tid] - lse;
    }
  }
}

// ---------------- Kernel 3: alpha wavefront DP + loss ----------------
__global__ void k_dp(const float* __restrict__ blankO, const float* __restrict__ emitO,
                     float* __restrict__ out) {
  __shared__ float buf[2][Bn][96];
  __shared__ float ll[Bn];
  const int tid = threadIdx.x;
  const int b = tid / 96, u = tid % 96;
  const bool active = (b < Bn) && (u <= Un);
  const float NEGINF = -INFINITY;
  float myPrev = NEGINF, bG = 0.f, eG = 0.f;

  for (int d = 0; d < Tn + Un; ++d) {
    const int p = (d & 1) ^ 1, c = d & 1;
    const int t = d - u;
    if (active && t >= 0 && t < Tn) {
      float val;
      if (d == 0) {
        val = 0.f;                                   // alpha[0][0]
      } else {
        float aa = (t > 0) ? myPrev + bG : NEGINF;
        float ee = (u > 0) ? buf[p][b][u - 1] + eG : NEGINF;
        float mx = fmaxf(aa, ee);
        val = (mx == NEGINF) ? NEGINF
                             : mx + log1pf(__expf(fminf(aa, ee) - mx));
      }
      myPrev = val;
      buf[c][b][u] = val;
    }
    // prefetch operands for next diagonal (hidden behind the barrier)
    int t2 = d + 1 - u;
    if (active && t2 >= 1 && t2 < Tn) bG = blankO[(b * Tn + t2 - 1) * (Un + 1) + u];
    if (active && t2 >= 0 && t2 < Tn && u > 0) eG = emitO[(b * Tn + t2) * Un + (u - 1)];
    __syncthreads();
  }

  if (active && u == Un)
    ll[b] = myPrev + blankO[(b * Tn + (Tn - 1)) * (Un + 1) + Un];
  __syncthreads();
  if (tid == 0) out[0] = -(ll[0] + ll[1] + ll[2] + ll[3]) * 0.25f;
}

// ---------------- launcher ----------------
extern "C" void kernel_launch(void* const* d_in, const int* in_sizes, int n_in,
                              void* d_out, int out_size, void* d_ws, size_t ws_size,
                              hipStream_t stream) {
  const float* enc = (const float*)d_in[0];
  const float* dec = (const float*)d_in[1];
  const float* Wf  = (const float*)d_in[2];
  const float* bfv = (const float*)d_in[3];
  const float* Wp  = (const float*)d_in[4];
  const float* bp  = (const float*)d_in[5];
  const int*   tg  = (const int*)d_in[6];
  float* out = (float*)d_out;

  char* ws = (char*)d_ws;
  float*  encP   = (float*)(ws + 0);          // 1200*512*4 = 2,457,600
  float*  decP   = (float*)(ws + 2457600);    // 324*512*4  =   663,552
  bf16_t* WpB    = (bf16_t*)(ws + 3121152);   // 1024*512*2 = 1,048,576
  float*  blankO = (float*)(ws + 4169728);    // 1200*81*4  =   388,800
  float*  emitO  = (float*)(ws + 4558528);    // 1200*80*4  =   384,000

  k_wp_bf16<<<(Vn * IN_) / 256, 256, 0, stream>>>(Wp, WpB);
  k_proj<<<((Bn * Tn + Bn * (Un + 1)) * IN_) / 256, 256, 0, stream>>>(enc, dec, Wf, bfv, encP, decP);
  k_joint<<<Bn * Tn * 3, 256, 39936, stream>>>(encP, decP, WpB, Wp, bp, tg, blankO, emitO);
  k_dp<<<1, 384, 0, stream>>>(blankO, emitO, out);
}
